// SelfCorrelationComputation10_584115552559
// MI455X (gfx1250) — compile-verified, matrix operand order checked by reading, not run on hardware
//
#include <hip/hip_runtime.h>
#include <hip/hip_bf16.h>

typedef _Float16 v16h __attribute__((ext_vector_type(16)));
typedef _Float16 v8h  __attribute__((ext_vector_type(8)));
typedef float    v8f  __attribute__((ext_vector_type(8)));

#define BATCH   8
#define CH      640
#define HH      40
#define WW      40
#define NPIX    (HH * WW)            // 1600
#define MTOT    (BATCH * NPIX)       // 12800
#define HEADS   8
#define HD      (CH / HEADS)         // 80
#define QKV_N   (3 * CH)             // 1920
#define SCALE   0.11180339887498948f // 80^-0.5

// ---------------------------------------------------------------------------
// CDNA5 async global->LDS copy (16B per lane), tracked by ASYNCcnt.
// vdst = wave-relative LDS byte address (low 32 bits of generic LDS pointer).
// ---------------------------------------------------------------------------
__device__ __forceinline__ void async_copy_b128(unsigned int lds_addr,
                                                const void* gaddr) {
    asm volatile("global_load_async_to_lds_b128 %0, %1, off"
                 :: "v"(lds_addr), "v"(gaddr)
                 : "memory");
}
__device__ __forceinline__ void wait_async_le4() {
    asm volatile("s_wait_asynccnt 0x4" ::: "memory");
}
__device__ __forceinline__ void wait_async_le0() {
    asm volatile("s_wait_asynccnt 0x0" ::: "memory");
}

// ---------------------------------------------------------------------------
// Kernel 1: ReLU + L2-normalize over C, NCHW -> (B*N) x C row-major f16
// ---------------------------------------------------------------------------
__global__ void prep_x(const float* __restrict__ x, _Float16* __restrict__ Xh) {
    int p = blockIdx.x * blockDim.x + threadIdx.x;      // 0..MTOT-1
    int b = p / NPIX, n = p % NPIX;
    const float* xb = x + (size_t)b * CH * NPIX + n;    // stride NPIX over c
    float ss = 0.f;
    for (int c = 0; c < CH; ++c) {
        float v = xb[(size_t)c * NPIX];
        v = v > 0.f ? v : 0.f;
        ss += v * v;
    }
    float denom = fmaxf(sqrtf(ss), 1e-12f);
    float inv = 1.f / denom;
    _Float16* o = Xh + (size_t)p * CH;
    for (int c0 = 0; c0 < CH; c0 += 8) {
        v8h pack;
#pragma unroll
        for (int u = 0; u < 8; ++u) {
            float v = xb[(size_t)(c0 + u) * NPIX];
            v = v > 0.f ? v : 0.f;
            pack[u] = (_Float16)(v * inv);
        }
        *reinterpret_cast<v8h*>(o + c0) = pack;   // 16B store
    }
}

// ---------------------------------------------------------------------------
// Kernel 2: weight conversion f32 -> f16 (qkv_w then proj_w)
// ---------------------------------------------------------------------------
__global__ void cvt_w(const float* __restrict__ qw, const float* __restrict__ pw,
                      _Float16* __restrict__ Wh, _Float16* __restrict__ Ph) {
    int i = blockIdx.x * blockDim.x + threadIdx.x;
    const int nq = QKV_N * CH;
    const int np = CH * CH;
    if (i < nq) {
        Wh[i] = (_Float16)qw[i];
    } else if (i < nq + np) {
        Ph[i - nq] = (_Float16)pw[i - nq];
    }
}

// ---------------------------------------------------------------------------
// WMMA GEMM: C[M,N] = A[M,K] * W[N,K]^T   (K = 640, f16 in, f32 acc)
// Block tile 128x128, 8 waves (2x4), wave tile 64x32 = 4x2 WMMA 16x16 accs.
// Double-buffered LDS staging via global_load_async_to_lds_b128 (ASYNCcnt).
// mode 0: QKV epilogue (bias, split q/k/v, scale q)
// mode 1: proj epilogue (bias, transposed NCHW f32 store)
// ---------------------------------------------------------------------------
#define LDSK   40               // padded K stride (f16): 80B rows, 16B chunks
#define TILEH  (128 * LDSK)     // elems per buffer (5120)
#define BUFBYT (TILEH * 2)      // bytes per buffer (10240)
#define KT     (CH / 32)        // 20 K-steps

__device__ __forceinline__ v16h load_frag(const _Float16* lds, int row, int kh) {
    const v8h* p0 = reinterpret_cast<const v8h*>(lds + row * LDSK + kh * 8);
    const v8h* p1 = reinterpret_cast<const v8h*>(lds + row * LDSK + 16 + kh * 8);
    v8h lo = *p0;
    v8h hi = *p1;
    return __builtin_shufflevector(lo, hi, 0, 1, 2, 3, 4, 5, 6, 7,
                                   8, 9, 10, 11, 12, 13, 14, 15);
}

__global__ __launch_bounds__(256) void gemm_wmma(
    const _Float16* __restrict__ A, const _Float16* __restrict__ Bw,
    const float* __restrict__ bias, int mode,
    float* __restrict__ qb, float* __restrict__ kb, float* __restrict__ vb,
    float* __restrict__ outp) {

    __shared__ __align__(16) _Float16 sA[2 * TILEH];
    __shared__ __align__(16) _Float16 sB[2 * TILEH];

    const int tid  = threadIdx.x;
    const int wid  = tid >> 5;
    const int lane = tid & 31;
    const int lm   = lane & 15;   // M (or N) row within fragment
    const int kh   = lane >> 4;   // K half selector
    const int wm   = wid >> 2;    // 0..1
    const int wn   = wid & 3;     // 0..3
    const int m0   = blockIdx.x * 128;
    const int n0   = blockIdx.y * 128;

    // Per-thread staging job: 2 chunks of 8 f16 for A and for B.
    const int row0 = tid >> 2;           // 0..63
    const int kc   = (tid & 3) * 8;      // 0,8,16,24
    const _Float16* gA0 = A  + (size_t)(m0 + row0) * CH + kc;
    const _Float16* gA1 = A  + (size_t)(m0 + row0 + 64) * CH + kc;
    const _Float16* gB0 = Bw + (size_t)(n0 + row0) * CH + kc;
    const _Float16* gB1 = Bw + (size_t)(n0 + row0 + 64) * CH + kc;
    const unsigned int lA0 = (unsigned int)(size_t)&sA[row0 * LDSK + kc];
    const unsigned int lB0 = (unsigned int)(size_t)&sB[row0 * LDSK + kc];
    const unsigned int rowHalf = 64 * LDSK * 2;   // +64 rows in bytes

    v8f acc[4][2] = {};

    // Prologue: async-stage tile 0 into buffer 0.
    async_copy_b128(lA0,           gA0);
    async_copy_b128(lA0 + rowHalf, gA1);
    async_copy_b128(lB0,           gB0);
    async_copy_b128(lB0 + rowHalf, gB1);

    for (int kt = 0; kt < KT; ++kt) {
        const int cur = kt & 1;
        if (kt + 1 < KT) {
            // Issue next tile into the other buffer (it was last read one
            // iteration ago; the trailing barrier of iter kt-1 fenced it).
            const unsigned int bo = (cur ^ 1) * (unsigned int)BUFBYT;
            const int ko = (kt + 1) * 32;
            async_copy_b128(lA0 + bo,           gA0 + ko);
            async_copy_b128(lA0 + bo + rowHalf, gA1 + ko);
            async_copy_b128(lB0 + bo,           gB0 + ko);
            async_copy_b128(lB0 + bo + rowHalf, gB1 + ko);
            wait_async_le4();   // loads complete in order: current tile done
        } else {
            wait_async_le0();
        }
        __syncthreads();        // everyone's current-tile writes visible

        const _Float16* curA = sA + cur * TILEH;
        const _Float16* curB = sB + cur * TILEH;
        v16h af[4], bf[2];
#pragma unroll
        for (int i = 0; i < 4; ++i)
            af[i] = load_frag(curA, wm * 64 + i * 16 + lm, kh);
#pragma unroll
        for (int j = 0; j < 2; ++j)
            bf[j] = load_frag(curB, wn * 32 + j * 16 + lm, kh);
#pragma unroll
        for (int i = 0; i < 4; ++i)
#pragma unroll
            for (int j = 0; j < 2; ++j)
                acc[i][j] = __builtin_amdgcn_wmma_f32_16x16x32_f16(
                    false, af[i], false, bf[j], (short)0, acc[i][j], false, false);
        __syncthreads();        // done reading buf[cur] before it is refilled
    }

    // Epilogue
#pragma unroll
    for (int i = 0; i < 4; ++i) {
#pragma unroll
        for (int j = 0; j < 2; ++j) {
            int gm_base = m0 + wm * 64 + i * 16 + kh * 8;
            int gn      = n0 + wn * 32 + j * 16 + lm;
            float bval  = bias[gn];
            if (mode == 0) {
#pragma unroll
                for (int r = 0; r < 8; ++r) {
                    int   gm  = gm_base + r;
                    float val = acc[i][j][r] + bval;
                    if (gn < CH) {
                        qb[(size_t)gm * CH + gn] = val * SCALE;
                    } else if (gn < 2 * CH) {
                        kb[(size_t)gm * CH + (gn - CH)] = val;
                    } else {
                        vb[(size_t)gm * CH + (gn - 2 * CH)] = val;
                    }
                }
            } else {
#pragma unroll
                for (int r = 0; r < 8; ++r) {
                    int   gm  = gm_base + r;
                    float val = acc[i][j][r] + bval;
                    int   b   = gm / NPIX;
                    int   n   = gm - b * NPIX;
                    outp[((size_t)b * CH + gn) * NPIX + n] = val;
                }
            }
        }
    }
}

// ---------------------------------------------------------------------------
// Kernel 4: 3x3 edge-clamped neighborhood attention + relative position bias
// One thread per (b, n, head); hd = 80 (20 float4 chunks).
// ---------------------------------------------------------------------------
__global__ __launch_bounds__(256) void nat_attn(
    const float* __restrict__ q, const float* __restrict__ kbuf,
    const float* __restrict__ vbuf, const float* __restrict__ rpb,
    _Float16* __restrict__ ao) {

    int t = blockIdx.x * blockDim.x + threadIdx.x;  // < MTOT * HEADS
    int h = t & (HEADS - 1);
    int p = t >> 3;
    int b = p / NPIX, n = p % NPIX;
    int y = n / WW, xq = n % WW;

    const float4* qp = reinterpret_cast<const float4*>(q + (size_t)p * CH + h * HD);
    float4 qr[HD / 4];
#pragma unroll
    for (int d = 0; d < HD / 4; ++d) qr[d] = qp[d];

    int cy = y < 1 ? 1 : (y > HH - 2 ? HH - 2 : y);
    int cx = xq < 1 ? 1 : (xq > WW - 2 ? WW - 2 : xq);
    // bias base row/col per _bias_idx (reversed flat index => 2 - ih pattern)
    int br = (y == 0) ? 2 : ((y == HH - 1) ? 0 : 1);
    int bc = (xq == 0) ? 2 : ((xq == WW - 1) ? 0 : 1);

    float logits[9];
    int base_b = b * NPIX;
#pragma unroll
    for (int di = 0; di < 3; ++di) {
#pragma unroll
        for (int dj = 0; dj < 3; ++dj) {
            int npix = base_b + (cy - 1 + di) * WW + (cx - 1 + dj);
            const float4* kp =
                reinterpret_cast<const float4*>(kbuf + (size_t)npix * CH + h * HD);
            float s = 0.f;
#pragma unroll
            for (int d = 0; d < HD / 4; ++d) {
                float4 kv = kp[d];
                s += qr[d].x * kv.x + qr[d].y * kv.y + qr[d].z * kv.z + qr[d].w * kv.w;
            }
            logits[di * 3 + dj] =
                s + rpb[h * 25 + (br + di) * 5 + (bc + dj)];
        }
    }

    float mx = logits[0];
#pragma unroll
    for (int kk = 1; kk < 9; ++kk) mx = fmaxf(mx, logits[kk]);
    float sum = 0.f;
#pragma unroll
    for (int kk = 0; kk < 9; ++kk) {
        logits[kk] = __expf(logits[kk] - mx);
        sum += logits[kk];
    }
    float inv = 1.f / sum;

    float4 o[HD / 4];
#pragma unroll
    for (int d = 0; d < HD / 4; ++d) o[d] = make_float4(0.f, 0.f, 0.f, 0.f);
#pragma unroll
    for (int di = 0; di < 3; ++di) {
#pragma unroll
        for (int dj = 0; dj < 3; ++dj) {
            int npix = base_b + (cy - 1 + di) * WW + (cx - 1 + dj);
            float w = logits[di * 3 + dj] * inv;
            const float4* vp =
                reinterpret_cast<const float4*>(vbuf + (size_t)npix * CH + h * HD);
#pragma unroll
            for (int d = 0; d < HD / 4; ++d) {
                float4 vv = vp[d];
                o[d].x += w * vv.x;
                o[d].y += w * vv.y;
                o[d].z += w * vv.z;
                o[d].w += w * vv.w;
            }
        }
    }

    _Float16* op = ao + (size_t)p * CH + h * HD;
#pragma unroll
    for (int d = 0; d < HD / 4; ++d) {
        op[4 * d + 0] = (_Float16)o[d].x;
        op[4 * d + 1] = (_Float16)o[d].y;
        op[4 * d + 2] = (_Float16)o[d].z;
        op[4 * d + 3] = (_Float16)o[d].w;
    }
}

// ---------------------------------------------------------------------------
// Launch
// ---------------------------------------------------------------------------
extern "C" void kernel_launch(void* const* d_in, const int* in_sizes, int n_in,
                              void* d_out, int out_size, void* d_ws, size_t ws_size,
                              hipStream_t stream) {
    const float* x      = (const float*)d_in[0];
    const float* qkv_w  = (const float*)d_in[1];
    const float* qkv_b  = (const float*)d_in[2];
    const float* proj_w = (const float*)d_in[3];
    const float* proj_b = (const float*)d_in[4];
    const float* rpb    = (const float*)d_in[5];
    float*       out    = (float*)d_out;

    char* ws = (char*)d_ws;
    size_t off = 0;
    _Float16* Xh = (_Float16*)(ws + off); off += (size_t)MTOT * CH * 2;   // 16.38 MB
    _Float16* Wh = (_Float16*)(ws + off); off += (size_t)QKV_N * CH * 2;  //  2.46 MB
    _Float16* Ph = (_Float16*)(ws + off); off += (size_t)CH * CH * 2;     //  0.82 MB
    float*    qb = (float*)(ws + off);    off += (size_t)MTOT * CH * 4;   // 32.77 MB
    float*    kb = (float*)(ws + off);    off += (size_t)MTOT * CH * 4;   // 32.77 MB
    float*    vb = (float*)(ws + off);    off += (size_t)MTOT * CH * 4;   // 32.77 MB
    _Float16* AO = (_Float16*)(ws + off); off += (size_t)MTOT * CH * 2;   // 16.38 MB

    // 1) preprocess: relu + l2norm + transpose + f16
    prep_x<<<MTOT / 256, 256, 0, stream>>>(x, Xh);

    // 2) weight conversion
    int ncvt = QKV_N * CH + CH * CH;
    cvt_w<<<(ncvt + 255) / 256, 256, 0, stream>>>(qkv_w, proj_w, Wh, Ph);

    // 3) QKV GEMM (12800 x 1920 x 640)
    gemm_wmma<<<dim3(MTOT / 128, QKV_N / 128), 256, 0, stream>>>(
        Xh, Wh, qkv_b, 0, qb, kb, vb, nullptr);

    // 4) neighborhood attention
    nat_attn<<<(MTOT * HEADS) / 256, 256, 0, stream>>>(qb, kb, vb, rpb, AO);

    // 5) projection GEMM (12800 x 640 x 640) + transposed NCHW store
    gemm_wmma<<<dim3(MTOT / 128, CH / 128), 256, 0, stream>>>(
        AO, Ph, proj_b, 1, nullptr, nullptr, nullptr, out);
}
